// FFTLayer_72670846648963
// MI455X (gfx1250) — matrix operand order repaired and checked by reading, not compiled, not run on hardware
//
#include <hip/hip_runtime.h>
#include <hip/hip_bf16.h>

// ---------------- problem constants ----------------
#define Bz   64
#define Cc   64
#define Lx   8192
#define Mm   4096            // half-size complex FFT length
#define Ff   4097            // rfft bins
#define FPAD 4112            // padded F stride (multiple of 16)
#define Hh   128             // CnnEncoder hidden = 2*C
#define Nn   3
#define CKk  192             // C*K im2col depth
#define NTILE 257            // ceil(4097/16)
#define NSPLIT 8
#define TPS   33             // tiles per split (8*33 >= 257)
#define TWN  2080            // 2048 twiddles + bank padding

typedef __attribute__((ext_vector_type(16))) __bf16 v16bf;
typedef __attribute__((ext_vector_type(8)))  float  v8f;

union BF16x16 {
    v16bf v;
    unsigned short u[16];
    unsigned int   d[8];
};

__device__ __forceinline__ unsigned short f32_to_bf16(float f) {
    unsigned int u = __float_as_uint(f);
    u += 0x7FFFu + ((u >> 16) & 1u);     // round-to-nearest-even
    return (unsigned short)(u >> 16);
}

__device__ __forceinline__ float selu_f(float x) {
    const float a  = 1.6732632423543772f;
    const float sc = 1.0507009873554805f;
    return sc * (x > 0.f ? x : a * (__expf(x) - 1.f));
}

__device__ __forceinline__ int rev12(int i) { return (int)(__brev((unsigned)i) >> 20); }

// ---------------- twiddle table: W_4096^p = exp(sign*2pi*i*p/4096) ----------------
__device__ __forceinline__ void fill_tw(float* twr, float* twi, int tid, float sign) {
    for (int it = 0; it < 8; ++it) {
        int p = it * 256 + tid;                       // 0..2047
        float ang = sign * 6.283185307179586f * (float)p * (1.0f / 4096.0f);
        float sv, cv;
        __sincosf(ang, &sv, &cv);
        int pp = p + (p >> 6);                        // bank-conflict padding
        twr[pp] = cv;
        twi[pp] = sv;
    }
}

// ---------------- in-LDS radix-2 FFT core (4096 pts, 256 threads) ----------------
__device__ __forceinline__ void fft4096(float* re, float* im,
                                        const float* twr, const float* twi, int tid) {
    for (int s = 1; s <= 12; ++s) {
        int half = 1 << (s - 1);
        int sh = 12 - s;
        for (int it = 0; it < 8; ++it) {
            int j   = it * 256 + tid;                 // 0..2047 butterflies
            int pos = j & (half - 1);
            int i0  = ((j >> (s - 1)) << s) + pos;
            int i1  = i0 + half;
            int p   = pos << sh;
            int pp  = p + (p >> 6);
            float wr = twr[pp], wi = twi[pp];
            float ar = re[i1], ai = im[i1];
            float tr = wr * ar - wi * ai;
            float ti = wr * ai + wi * ar;
            float br = re[i0], bi = im[i0];
            re[i0] = br + tr; im[i0] = bi + ti;
            re[i1] = br - tr; im[i1] = bi - ti;
        }
        __syncthreads();
    }
}

// ---------------- kernel 1: forward rfft (ortho) via packed half-size FFT ----------------
__global__ __launch_bounds__(256)
void fft_forward_kernel(const float* __restrict__ x,
                        float* __restrict__ fxr, float* __restrict__ fxi) {
    __shared__ float re[Mm], im[Mm], twr[TWN], twi[TWN];
    int tid = threadIdx.x;
    size_t row = (size_t)blockIdx.x * (size_t)Lx;
    fill_tw(twr, twi, tid, -1.f);
    // z[n] = x[2n] + i*x[2n+1], scattered bit-reversed
    for (int it = 0; it < 16; ++it) {
        int n = it * 256 + tid;                       // 0..4095
        float2 v = *(const float2*)(x + row + 2 * n);
        int rv = rev12(n);
        re[rv] = v.x;
        im[rv] = v.y;
    }
    __syncthreads();
    fft4096(re, im, twr, twi, tid);
    // split/merge: X[k] = E[k] + W_8192^k * O[k]
    const float sc = 0.011048543456039805f;           // 1/sqrt(8192)
    const float twoPiN = 6.283185307179586f / 8192.f;
    size_t frow = (size_t)blockIdx.x * (size_t)FPAD;
    for (int it = 0; it < 9; ++it) {
        int k = it * 256 + tid;
        if (k <= 2048) {
            int a  = k & (Mm - 1);
            int b2 = (Mm - k) & (Mm - 1);
            float Zra = re[a], Zia = im[a], Zrb = re[b2], Zib = im[b2];
            float Er =  0.5f * (Zra + Zrb);
            float Ei =  0.5f * (Zia - Zib);
            float Or =  0.5f * (Zia + Zib);
            float Oi = -0.5f * (Zra - Zrb);
            float sv, cv;
            __sincosf(twoPiN * (float)k, &sv, &cv);   // W = cv - i*sv
            fxr[frow + k] = (Er + cv * Or + sv * Oi) * sc;
            fxi[frow + k] = (Ei + cv * Oi - sv * Or) * sc;
            if (k <= 2047) {                          // X[4096-k] = conj(E) - (cv+i*sv)*conj(O)
                fxr[frow + (Mm - k)] = (Er - (cv * Or + sv * Oi)) * sc;
                fxi[frow + (Mm - k)] = (-Ei + cv * Oi + sv * Or) * sc;
            }
        }
    }
    // zero pad region [4097,4112) so conv staging can load unguarded
    if (tid < FPAD - Ff) {
        fxr[frow + Ff + tid] = 0.f;
        fxi[frow + Ff + tid] = 0.f;
    }
}

// ---------------- kernel 2: conv1+SELU+pooled-conv2 stats via bf16 WMMA ----------------
// dynamic LDS: [ Xt buf0 | Xt buf1 | Bt ]
// Xt row layout (stride 24 floats): slot 3 = f0-1 edge, slots 4..19 = f0..f0+15 (16B
// aligned for async b128), slot 20 = f0+16 halo. im2col read index = n + k3 + 3.
#define XTROW      24
#define XTBUF_BYTES (Cc * XTROW * 4)          // 6144
#define BT_OFF      (2 * XTBUF_BYTES)         // 12288
#define BT_BYTES    (6 * 2 * 16 * 16 * 2)     // 6144
#define SMEM_TOTAL  (BT_OFF + BT_BYTES)       // 18432

// async staging of one 16-column slab (+halo) for the caller's channel
__device__ __forceinline__ void stage_async(const float* Xc, int cch, int q4,
                                            int f0, unsigned bufBase) {
    // 4 lanes/channel: each copies 16B (slots 4+4q .. 7+4q  <-  f0+4q .. f0+4q+3)
    unsigned lds128 = bufBase + (unsigned)((cch * XTROW + 4 + q4 * 4) * 4);
    unsigned long long ga = (unsigned long long)(Xc + f0 + q4 * 4);
    asm volatile("global_load_async_to_lds_b128 %0, %1, off"
                 :: "v"(lds128), "v"(ga) : "memory");
    if (q4 == 1) {                                    // halo column f0+16 -> slot 20
        unsigned lds32 = bufBase + (unsigned)((cch * XTROW + 20) * 4);
        unsigned long long gb = (unsigned long long)(Xc + f0 + 16);
        asm volatile("global_load_async_to_lds_b32 %0, %1, off"
                     :: "v"(lds32), "v"(gb) : "memory");
    }
}

__global__ __launch_bounds__(256)
void conv_enc_kernel(const float* __restrict__ fxr, const float* __restrict__ fxi,
                     const float* __restrict__ wR1, const float* __restrict__ bR1,
                     const float* __restrict__ wI1, const float* __restrict__ bI1,
                     float* __restrict__ Ssum, float* __restrict__ Fv,
                     float* __restrict__ Lv) {
    extern __shared__ char smem[];

    const int b = blockIdx.x, e = blockIdx.y, split = blockIdx.z;
    const float* X  = (e == 0 ? fxr : fxi) + (size_t)b * Cc * FPAD;
    const float* W1 = (e == 0 ? wR1 : wI1);
    const float* B1 = (e == 0 ? bR1 : bI1);

    const int tid  = threadIdx.x;
    const int lane = tid & 31;
    const int wv   = tid >> 5;          // 8 waves
    const int h0   = wv * 16;           // 16 output channels per wave
    const int hi   = lane >> 4;
    const int nl   = lane & 15;
    const int cch  = tid >> 2;          // staging channel for this thread
    const int q4   = tid & 3;
    const float* Xc = X + (size_t)cch * FPAD;

    // kick off the first tile's async staging before the weight preload
    const int tile0 = split * TPS;
    if (tile0 < NTILE) stage_async(Xc, cch, q4, tile0 << 4, 0u);

    // Preload A-matrix (weights) into registers in 16x32 bf16 layout, contiguous k = 16*hi + t.
    BF16x16 Areg[6];
    const int arow = h0 + nl;
    for (int q = 0; q < 6; ++q)
        for (int t = 0; t < 16; ++t) {
            int kk = hi * 16 + t;
            Areg[q].u[t] = f32_to_bf16(W1[arow * CKk + q * 32 + kk]);
        }
    float bias[8];
    for (int i = 0; i < 8; ++i) bias[i] = B1[h0 + hi * 8 + i];

    float acc[8], firstv[8], lastv[8];
    for (int i = 0; i < 8; ++i) { acc[i] = 0.f; firstv[i] = 0.f; lastv[i] = 0.f; }

    unsigned curBase = 0;
    for (int tl = 0; tl < TPS; ++tl) {
        int tile = split * TPS + tl;
        if (tile >= NTILE) break;
        int f0 = tile << 4;
        float* Xtc = (float*)(smem + curBase);
        // wait for this tile's async slab, patch the guarded f0-1 edge
        asm volatile("s_wait_asynccnt 0x0" ::: "memory");
        if (q4 == 0) Xtc[cch * XTROW + 3] = (f0 > 0) ? Xc[f0 - 1] : 0.f;
        __syncthreads();
        // ---- build im2col B tile (192 x 16) in WMMA B lane layout, packed b32 stores ----
        for (int ii = 0; ii < 6; ++ii) {
            int e2 = ii * 256 + tid;           // dword index 0..1535
            int eI = e2 * 2;
            int q  = eI >> 9;
            int r  = eI & 511;
            int hf = r >> 8;
            int n  = (r >> 4) & 15;
            int t  = r & 15;                   // even
            int ck0 = eI >= 3072 ? 0 : (q * 32 + hf * 16 + t);
            ck0 = q * 32 + hf * 16 + t;
            int c0  = ck0 / 3;          int k30 = ck0 - c0 * 3;
            int c1  = (ck0 + 1) / 3;    int k31 = (ck0 + 1) - c1 * 3;
            unsigned lo  = f32_to_bf16(Xtc[c0 * XTROW + n + k30 + 3]);
            unsigned hi2 = f32_to_bf16(Xtc[c1 * XTROW + n + k31 + 3]);
            ((unsigned*)(smem + BT_OFF))[e2] = lo | (hi2 << 16);
        }
        __syncthreads();
        // ---- prefetch next tile into the other buffer (overlaps the WMMAs) ----
        if (tl + 1 < TPS && tile + 1 < NTILE)
            stage_async(Xc, cch, q4, (tile + 1) << 4, curBase ^ (unsigned)XTBUF_BYTES);
        // ---- 6 chained WMMAs: [16h x 192] x [192 x 16f] ----
        v8f cacc = {};
        const uint4* BtV = (const uint4*)(smem + BT_OFF);
        for (int q = 0; q < 6; ++q) {
            BF16x16 Bv;
            int base = ((q * 2 + hi) * 16 + nl) * 2;
            uint4 lo4 = BtV[base + 0];
            uint4 hi4 = BtV[base + 1];
            Bv.d[0] = lo4.x; Bv.d[1] = lo4.y; Bv.d[2] = lo4.z; Bv.d[3] = lo4.w;
            Bv.d[4] = hi4.x; Bv.d[5] = hi4.y; Bv.d[6] = hi4.z; Bv.d[7] = hi4.w;
            cacc = __builtin_amdgcn_wmma_f32_16x16x32_bf16(
                false, Areg[q].v, false, Bv.v, (short)0, cacc, false, false);
        }
        // ---- bias + SELU + masked column-sum + edge capture ----
        int colf = f0 + nl;
        bool valid = colf < Ff;
        for (int i = 0; i < 8; ++i) {
            float s = selu_f(cacc[i] + bias[i]);
            if (valid) acc[i] += s;
            if (f0 == 0 && nl == 0)    firstv[i] = s;   // y1[h][0]
            if (colf == Ff - 1)        lastv[i]  = s;   // y1[h][F-1]
        }
        curBase ^= (unsigned)XTBUF_BYTES;
    }
    // reduce column sums across the 16 lanes sharing each row
    for (int i = 0; i < 8; ++i) {
        float a = acc[i];
        a += __shfl_xor(a, 1, 16);
        a += __shfl_xor(a, 2, 16);
        a += __shfl_xor(a, 4, 16);
        a += __shfl_xor(a, 8, 16);
        if (nl == 0) {
            int row = h0 + hi * 8 + i;
            size_t o = ((size_t)e * Bz + b) * Hh + row;
            atomicAdd(&Ssum[o], a);
            if (split == 0)          Fv[o] = firstv[i];
            if (split == NSPLIT - 1) Lv[o] = lastv[i];
        }
    }
}

// ---------------- kernel 2b: pooled conv2 + softmax -> C coefficients ----------------
__global__ __launch_bounds__(128)
void logits_kernel(const float* __restrict__ Ssum, const float* __restrict__ Fv,
                   const float* __restrict__ Lv,
                   const float* __restrict__ wR2, const float* __restrict__ bR2,
                   const float* __restrict__ wI2, const float* __restrict__ bI2,
                   float* __restrict__ Cenc) {
    int b = blockIdx.x, e = blockIdx.y, h = threadIdx.x;
    const float* W2 = (e == 0 ? wR2 : wI2);
    const float* B2 = (e == 0 ? bR2 : bI2);
    size_t o = ((size_t)e * Bz + b) * Hh + h;
    float s = Ssum[o], fv = Fv[o], lv = Lv[o];
    __shared__ float red[3][128];
    for (int n = 0; n < 3; ++n) {
        float w0 = W2[(n * Hh + h) * 3 + 0];
        float w1 = W2[(n * Hh + h) * 3 + 1];
        float w2 = W2[(n * Hh + h) * 3 + 2];
        red[n][h] = (w0 + w1 + w2) * s - w0 * lv - w2 * fv;  // SAME-pad edge algebra
    }
    __syncthreads();
    for (int st = 64; st > 0; st >>= 1) {
        if (h < st)
            for (int n = 0; n < 3; ++n) red[n][h] += red[n][h + st];
        __syncthreads();
    }
    if (h == 0) {
        const float invF = 1.0f / (float)Ff;
        float lg[3];
        for (int n = 0; n < 3; ++n) lg[n] = red[n][0] * invF + B2[n];
        float m = fmaxf(lg[0], fmaxf(lg[1], lg[2]));
        float e0 = __expf(lg[0] - m), e1 = __expf(lg[1] - m), e2 = __expf(lg[2] - m);
        float inv = 1.f / (e0 + e1 + e2);
        float* Cb = Cenc + ((size_t)e * Bz + b) * 3;
        Cb[0] = e0 * inv; Cb[1] = e1 * inv; Cb[2] = e2 * inv;
    }
}

// ---------------- kernel 3: Kf mix + complex mult + packed irfft (ortho) ----------------
__device__ __forceinline__ void computeG(const float* __restrict__ fxr,
                                         const float* __restrict__ fxi,
                                         const float* __restrict__ param,
                                         size_t frow, int c,
                                         const float* CR, const float* CI,
                                         int f, float& gr, float& gi) {
    float kr = 0.f, ki = 0.f;
    for (int n = 0; n < 3; ++n) {
        size_t pidx = (((size_t)n * Cc + c) * (size_t)Ff + f) * 2;
        kr += CR[n] * param[pidx];
        ki += CI[n] * param[pidx + 1];
    }
    float fr = fxr[frow + f], fi = fxi[frow + f];
    gr = fr * kr - fi * ki;
    gi = fr * ki + fi * kr;
}

__global__ __launch_bounds__(256)
void fft_inverse_kernel(const float* __restrict__ fxr, const float* __restrict__ fxi,
                        const float* __restrict__ param, const float* __restrict__ Cenc,
                        float* __restrict__ out) {
    __shared__ float re[Mm], im[Mm], twr[TWN], twi[TWN];
    int tid = threadIdx.x;
    int rowid = blockIdx.x;
    int b = rowid >> 6;
    int c = rowid & 63;
    float CR[3], CI[3];
    for (int n = 0; n < 3; ++n) {
        CR[n] = Cenc[b * 3 + n];
        CI[n] = Cenc[(Bz + b) * 3 + n];
    }
    fill_tw(twr, twi, tid, +1.f);
    size_t frow = (size_t)rowid * (size_t)FPAD;
    const float twoPiN = 6.283185307179586f / 8192.f;
    // build Z[k] = E[k] + i*O[k], scattered bit-reversed (fuses Hermitian ext + permute)
    for (int it = 0; it < 9; ++it) {
        int k = it * 256 + tid;
        if (k <= 2048) {
            float g1r, g1i, g2r, g2i;
            computeG(fxr, fxi, param, frow, c, CR, CI, k,       g1r, g1i);
            computeG(fxr, fxi, param, frow, c, CR, CI, Mm - k,  g2r, g2i);
            if (k == 0) { g1i = 0.f; g2i = 0.f; }     // c2r drops DC/Nyquist imag
            float Er = 0.5f * (g1r + g2r);
            float Ei = 0.5f * (g1i - g2i);
            float Wr = 0.5f * (g1r - g2r);
            float Wi = 0.5f * (g1i + g2i);
            float sv, cv;
            __sincosf(twoPiN * (float)k, &sv, &cv);   // e^{+i*theta}
            float Or = cv * Wr - sv * Wi;
            float Oi = cv * Wi + sv * Wr;
            int zi = rev12(k & (Mm - 1));
            re[zi] = Er - Oi;                         // Z[k] = E + i*O
            im[zi] = Ei + Or;
            if (k >= 1 && k <= 2047) {
                int zj = rev12(Mm - k);               // Z[M-k] = conj(E) + i*conj(O)
                re[zj] = Er + Oi;
                im[zj] = Or - Ei;
            }
        }
    }
    __syncthreads();
    fft4096(re, im, twr, twi, tid);
    // z[n] = x[2n] + i*x[2n+1]
    const float sc = 0.011048543456039805f;           // 1/sqrt(8192) ortho
    size_t orow = (size_t)rowid * (size_t)Lx;
    for (int it = 0; it < 16; ++it) {
        int n = it * 256 + tid;
        float2 v;
        v.x = re[n] * sc;
        v.y = im[n] * sc;
        *(float2*)(out + orow + 2 * n) = v;
    }
}

// ---------------- host-side launcher ----------------
extern "C" void kernel_launch(void* const* d_in, const int* in_sizes, int n_in,
                              void* d_out, int out_size, void* d_ws, size_t ws_size,
                              hipStream_t stream) {
    const float* x    = (const float*)d_in[0];
    const float* prm  = (const float*)d_in[1];
    const float* wR1  = (const float*)d_in[2];
    const float* bR1  = (const float*)d_in[3];
    const float* wR2  = (const float*)d_in[4];
    const float* bR2  = (const float*)d_in[5];
    const float* wI1  = (const float*)d_in[6];
    const float* bI1  = (const float*)d_in[7];
    const float* wI2  = (const float*)d_in[8];
    const float* bI2  = (const float*)d_in[9];
    float* out = (float*)d_out;
    float* ws  = (float*)d_ws;

    const size_t fxElems = (size_t)Bz * Cc * FPAD;
    float* fxr  = ws;
    float* fxi  = fxr + fxElems;
    float* Ssum = fxi + fxElems;                     // [2][B][H]
    float* Fv   = Ssum + (size_t)2 * Bz * Hh;        // y1[.,0]
    float* Lv   = Fv   + (size_t)2 * Bz * Hh;        // y1[.,F-1]
    float* Cenc = Lv   + (size_t)2 * Bz * Hh;        // [2][B][3]

    hipMemsetAsync(Ssum, 0,
                   ((size_t)3 * 2 * Bz * Hh + (size_t)2 * Bz * Nn) * sizeof(float),
                   stream);

    fft_forward_kernel<<<Bz * Cc, 256, 0, stream>>>(x, fxr, fxi);

    dim3 g2(Bz, 2, NSPLIT);
    conv_enc_kernel<<<g2, 256, SMEM_TOTAL, stream>>>(
        fxr, fxi, wR1, bR1, wI1, bI1, Ssum, Fv, Lv);

    dim3 g2b(Bz, 2);
    logits_kernel<<<g2b, 128, 0, stream>>>(Ssum, Fv, Lv, wR2, bR2, wI2, bI2, Cenc);

    fft_inverse_kernel<<<Bz * Cc, 256, 0, stream>>>(fxr, fxi, prm, Cenc, out);
}
